// FlatGCN_5798205849655
// MI455X (gfx1250) — matrix operand order repaired and checked by reading, not compiled, not run on hardware
//
#include <hip/hip_runtime.h>

#define NNODES 50000
#define NEDGES 800000
#define NTILES (NEDGES / 16)   // 50000
#define GTILES (NNODES / 16)   // 3125

typedef __attribute__((ext_vector_type(16))) _Float16 v16h;
typedef __attribute__((ext_vector_type(8)))  _Float16 v8h;
typedef __attribute__((ext_vector_type(8)))  float    v8f;
typedef __attribute__((ext_vector_type(4)))  float    v4f;

// Fragment region layout inside the packed-weight buffer (units of 512 halves):
//   frags   0..31 : gcn_w[0..3]           (8 frags per 64x64 matrix: nb*2+kc)
//   frags  32..35 : edge_w0 (K padded 16->32, 4 frags, one k-chunk)
//   frags  36..91 : edge_w[0..6]
//   frags 92..115 : out_w0 split rows [0:64)=x_src, [64:128)=x_dst, [128:192)=e
//   frags116..139 : out_w[0..2]
#define EDGE_FRAG_BASE 32
#define N_EDGE_FRAGS   108       // frags 32..139 relative base
#define TOTAL_FRAGS    140

__device__ __forceinline__ v8f wmma_f16(v16h a, v16h b, v8f c) {
  return __builtin_amdgcn_wmma_f32_16x16x32_f16(false, a, false, b, (short)0, c,
                                                false, false);
}

__device__ __forceinline__ v8f zero8() {
  v8f z;
#pragma unroll
  for (int j = 0; j < 8; ++j) z[j] = 0.0f;
  return z;
}

// B fragment: 16 consecutive halves per lane from packed buffer
__device__ __forceinline__ v16h ld_frag(const _Float16* p) {
  v8h lo = *(const v8h*)(p);
  v8h hi = *(const v8h*)(p + 8);
  return __builtin_shufflevector(lo, hi, 0,1,2,3,4,5,6,7,8,9,10,11,12,13,14,15);
}

// A fragment from an f16 row of 64 (LDS activation tile row, stride 72 halves,
// or a global f16 x-row, stride 64).  16-bit A layout: lane<16 -> K
// {k0+0..7, k0+16..23}; lane>=16 -> +8 offset.
__device__ __forceinline__ v16h ldA16(const _Float16* rowp, int k0, int koff) {
  v8h lo = *(const v8h*)(rowp + k0 + koff);
  v8h hi = *(const v8h*)(rowp + k0 + 16 + koff);
  return __builtin_shufflevector(lo, hi, 0,1,2,3,4,5,6,7,8,9,10,11,12,13,14,15);
}

// A fragment gathered from a global f32 row of 64, converted to f16
__device__ __forceinline__ v16h gA32(const float* rp, int k0, int koff) {
  v4f q0 = *(const v4f*)(rp + k0 + koff);
  v4f q1 = *(const v4f*)(rp + k0 + koff + 4);
  v4f q2 = *(const v4f*)(rp + k0 + 16 + koff);
  v4f q3 = *(const v4f*)(rp + k0 + 16 + koff + 4);
  v16h r;
#pragma unroll
  for (int j = 0; j < 4; ++j) {
    r[j]      = (_Float16)q0[j];
    r[4 + j]  = (_Float16)q1[j];
    r[8 + j]  = (_Float16)q2[j];
    r[12 + j] = (_Float16)q3[j];
  }
  return r;
}

// D (f32, col-striped) -> bias + relu -> f16 LDS activation tile
__device__ __forceinline__ void storeD_relu(_Float16* act, v8f acc,
                                            const float* __restrict__ bias,
                                            int nb, int lane) {
  const int n  = nb * 16 + (lane & 15);
  const int hi = lane >> 4;
  const float bv = bias[n];
#pragma unroll
  for (int j = 0; j < 8; ++j) {
    float v = acc[j] + bv;
    act[(hi * 8 + j) * 72 + n] = (_Float16)fmaxf(v, 0.0f);
  }
}

// ---------------- weight packing ----------------
__global__ void __launch_bounds__(256) k_pack(
    const float* __restrict__ gcn_w, const float* __restrict__ edge_w0,
    const float* __restrict__ edge_w, const float* __restrict__ out_w0,
    const float* __restrict__ out_w, _Float16* __restrict__ packed) {
  int gid = blockIdx.x * 256 + threadIdx.x;
  if (gid >= TOTAL_FRAGS * 512) return;
  const int frag  = gid >> 9;
  const int lane  = (gid >> 4) & 31;
  const int elem  = gid & 15;
  const int col16 = lane & 15;
  // B layout: lane<16 holds K {0..7, 16..23} of the 32-K chunk, lane>=16 holds K {8..15, 24..31}
  const int kk = ((lane & 16) ? 8 : 0) + ((elem & 8) ? 16 : 0) + (elem & 7);
  float val = 0.0f;
  if (frag < 32) {            // gcn_w
    int layer = frag >> 3, f = frag & 7, nb = f >> 1, kc = f & 1;
    val = gcn_w[layer * 4096 + (kc * 32 + kk) * 64 + (nb * 16 + col16)];
  } else if (frag < 36) {     // edge_w0, K padded 16->32
    int nb = frag - 32;
    if (kk < 16) val = edge_w0[kk * 64 + (nb * 16 + col16)];
  } else if (frag < 92) {     // edge_w
    int t = frag - 36, layer = t >> 3, f = t & 7, nb = f >> 1, kc = f & 1;
    val = edge_w[layer * 4096 + (kc * 32 + kk) * 64 + (nb * 16 + col16)];
  } else if (frag < 116) {    // out_w0 split into 3 row-blocks
    int t = frag - 92, p = t >> 3, f = t & 7, nb = f >> 1, kc = f & 1;
    val = out_w0[(p * 64 + kc * 32 + kk) * 64 + (nb * 16 + col16)];
  } else {                    // out_w
    int t = frag - 116, layer = t >> 3, f = t & 7, nb = f >> 1, kc = f & 1;
    val = out_w[layer * 4096 + (kc * 32 + kk) * 64 + (nb * 16 + col16)];
  }
  packed[gid] = (_Float16)val;
}

// ---------------- degree / norm ----------------
__global__ void __launch_bounds__(256) k_deg(const int* __restrict__ dst,
                                             float* __restrict__ dinv) {
  int g = blockIdx.x * 256 + threadIdx.x;
  if (g < NEDGES) atomicAdd(&dinv[dst[g]], 1.0f);
}

__global__ void __launch_bounds__(256) k_rsq(float* __restrict__ dinv) {
  int g = blockIdx.x * 256 + threadIdx.x;
  if (g < NNODES) dinv[g] = rsqrtf(dinv[g] + 1.0f);
}

// ---------------- GCN GEMM: xl = xin @ W (WMMA) ----------------
__global__ void __launch_bounds__(256) k_gcn_gemm(
    const float* __restrict__ xin, const _Float16* __restrict__ wpack,
    float* __restrict__ xl) {
  __shared__ _Float16 wl[8 * 512];
  {
    int t = threadIdx.x;
    for (int i = t; i < 512; i += 256)   // 8192 bytes
      ((int4*)wl)[i] = ((const int4*)wpack)[i];
  }
  __syncthreads();
  const int tid = threadIdx.x, wave = tid >> 5, lane = tid & 31;
  const int row = lane & 15, hi = lane >> 4, koff = hi * 8;
  for (int tile = blockIdx.x * 8 + wave; tile < GTILES; tile += gridDim.x * 8) {
    const float* rp = xin + ((size_t)tile * 16 + row) * 64;
    v16h a0 = gA32(rp, 0, koff);
    v16h a1 = gA32(rp, 32, koff);
#pragma unroll
    for (int nb = 0; nb < 4; ++nb) {
      v8f acc = zero8();
      acc = wmma_f16(a0, ld_frag(wl + (nb * 2 + 0) * 512 + lane * 16), acc);
      acc = wmma_f16(a1, ld_frag(wl + (nb * 2 + 1) * 512 + lane * 16), acc);
      const int n = nb * 16 + row;
#pragma unroll
      for (int j = 0; j < 8; ++j)
        xl[((size_t)tile * 16 + hi * 8 + j) * 64 + n] = acc[j];
    }
  }
}

// ---------------- scatter: agg[dst] += norm * xl[src] ----------------
__global__ void __launch_bounds__(256) k_scatter(
    const int* __restrict__ src, const int* __restrict__ dst,
    const float* __restrict__ dinv, const float* __restrict__ xl,
    float* __restrict__ agg) {
  unsigned gid = blockIdx.x * 256u + threadIdx.x;   // < E*64 = 51.2M
  int e = gid >> 6, c = gid & 63;
  int s = src[e], d = dst[e];
  float nrm = dinv[s] * dinv[d];
  atomicAdd(&agg[(size_t)d * 64 + c], nrm * xl[(size_t)s * 64 + c]);
}

// ---------------- x = relu(agg + self_norm*xl + b) ----------------
__global__ void __launch_bounds__(256) k_update(
    const float* __restrict__ agg, const float* __restrict__ xl,
    const float* __restrict__ dinv, const float* __restrict__ bias,
    float* __restrict__ xc) {
  unsigned gid = blockIdx.x * 256u + threadIdx.x;   // < N*64
  int i = gid >> 6, c = gid & 63;
  float di = dinv[i];
  xc[gid] = fmaxf(agg[gid] + di * di * xl[gid] + bias[c], 0.0f);
}

// ---------------- convert final node features to f16 once ----------------
__global__ void __launch_bounds__(256) k_cvt_x(const float* __restrict__ xc,
                                               _Float16* __restrict__ xh) {
  unsigned gid = blockIdx.x * 256u + threadIdx.x;   // < N*64
  xh[gid] = (_Float16)xc[gid];
}

// ---------------- fused per-edge MLP pipeline ----------------
__global__ void __launch_bounds__(256) k_edge_fused(
    const _Float16* __restrict__ xh,     // f16 node features [N,64]
    const int* __restrict__ src, const int* __restrict__ dst,
    const float* __restrict__ edge_attr,
    const _Float16* __restrict__ wpack,  // edge region base (108 frags)
    const float* __restrict__ edge_b0, const float* __restrict__ edge_b,
    const float* __restrict__ out_b0, const float* __restrict__ out_b,
    const float* __restrict__ fin_w, const float* __restrict__ fin_b,
    float* __restrict__ out) {
  extern __shared__ int4 smem4[];
  _Float16* wl     = (_Float16*)smem4;          // 108*512 = 55296 halves
  _Float16* actAll = wl + N_EDGE_FRAGS * 512;   // 8 waves * 1152 halves

  {
    int t = threadIdx.x;
    for (int i = t; i < 6912; i += 256)   // 110592 bytes of packed weights
      ((int4*)wl)[i] = ((const int4*)wpack)[i];
  }
  __syncthreads();

  const int tid = threadIdx.x, wave = tid >> 5, lane = tid & 31;
  const int row = lane & 15, hi = lane >> 4, koff = hi * 8;
  _Float16* act    = actAll + wave * 1152;      // 16 rows, stride 72 halves
  _Float16* actRow = act + row * 72;
  const float finb = fin_b[0];

  for (int tile = blockIdx.x * 8 + wave; tile < NTILES; tile += gridDim.x * 8) {
    const int e = tile * 16 + row;
    const int s = src[e], d = dst[e];

    // stage 0: e0 = relu(edge_attr @ edge_w0 + b0)   (K padded 16 -> 32)
    {
      const float* ap = edge_attr + (size_t)e * 16 + koff;
      v4f q0 = *(const v4f*)(ap);
      v4f q1 = *(const v4f*)(ap + 4);
      v16h a;
#pragma unroll
      for (int j = 0; j < 4; ++j) {
        a[j] = (_Float16)q0[j];  a[4 + j]  = (_Float16)q1[j];
        a[8 + j] = (_Float16)0.0f; a[12 + j] = (_Float16)0.0f;
      }
#pragma unroll
      for (int nb = 0; nb < 4; ++nb) {
        v8f acc = zero8();
        acc = wmma_f16(a, ld_frag(wl + nb * 512 + lane * 16), acc);
        storeD_relu(act, acc, edge_b0, nb, lane);
      }
    }
    __builtin_amdgcn_wave_barrier();

    // 7 hidden edge layers: e = relu(e @ edge_w[L] + b[L])
#pragma unroll 1
    for (int L = 0; L < 7; ++L) {
      const _Float16* base = wl + (size_t)(4 + L * 8) * 512;
      v16h a0 = ldA16(actRow, 0, koff);
      v16h a1 = ldA16(actRow, 32, koff);
      __builtin_amdgcn_wave_barrier();
#pragma unroll
      for (int nb = 0; nb < 4; ++nb) {
        v8f acc = zero8();
        acc = wmma_f16(a0, ld_frag(base + (nb * 2 + 0) * 512 + lane * 16), acc);
        acc = wmma_f16(a1, ld_frag(base + (nb * 2 + 1) * 512 + lane * 16), acc);
        storeD_relu(act, acc, edge_b + L * 64, nb, lane);
      }
      __builtin_amdgcn_wave_barrier();
    }

    // concat head: f = relu(x[src]@W0a + x[dst]@W0b + e@W0c + b)
    {
      v16h e0 = ldA16(actRow, 0, koff);
      v16h e1 = ldA16(actRow, 32, koff);
      const _Float16* xs = xh + (size_t)s * 64;
      const _Float16* xd = xh + (size_t)d * 64;
      v16h s0 = ldA16(xs, 0, koff), s1 = ldA16(xs, 32, koff);
      v16h d0 = ldA16(xd, 0, koff), d1 = ldA16(xd, 32, koff);
      __builtin_amdgcn_wave_barrier();
      const _Float16* bs = wl + (size_t)60 * 512;
      const _Float16* bd = wl + (size_t)68 * 512;
      const _Float16* be = wl + (size_t)76 * 512;
#pragma unroll
      for (int nb = 0; nb < 4; ++nb) {
        v8f acc = zero8();
        acc = wmma_f16(s0, ld_frag(bs + (nb * 2 + 0) * 512 + lane * 16), acc);
        acc = wmma_f16(s1, ld_frag(bs + (nb * 2 + 1) * 512 + lane * 16), acc);
        acc = wmma_f16(d0, ld_frag(bd + (nb * 2 + 0) * 512 + lane * 16), acc);
        acc = wmma_f16(d1, ld_frag(bd + (nb * 2 + 1) * 512 + lane * 16), acc);
        acc = wmma_f16(e0, ld_frag(be + (nb * 2 + 0) * 512 + lane * 16), acc);
        acc = wmma_f16(e1, ld_frag(be + (nb * 2 + 1) * 512 + lane * 16), acc);
        storeD_relu(act, acc, out_b0, nb, lane);
      }
      __builtin_amdgcn_wave_barrier();
    }

    // 3 out layers
#pragma unroll 1
    for (int L = 0; L < 3; ++L) {
      const _Float16* base = wl + (size_t)(84 + L * 8) * 512;
      v16h a0 = ldA16(actRow, 0, koff);
      v16h a1 = ldA16(actRow, 32, koff);
      __builtin_amdgcn_wave_barrier();
#pragma unroll
      for (int nb = 0; nb < 4; ++nb) {
        v8f acc = zero8();
        acc = wmma_f16(a0, ld_frag(base + (nb * 2 + 0) * 512 + lane * 16), acc);
        acc = wmma_f16(a1, ld_frag(base + (nb * 2 + 1) * 512 + lane * 16), acc);
        storeD_relu(act, acc, out_b + L * 64, nb, lane);
      }
      __builtin_amdgcn_wave_barrier();
    }

    // final: out = f @ fin_w + fin_b
    if (lane < 16) {
      const _Float16* r = act + lane * 72;
      float ssum = finb;
#pragma unroll
      for (int k = 0; k < 64; ++k) ssum += (float)r[k] * fin_w[k];
      out[tile * 16 + lane] = ssum;
    }
    __builtin_amdgcn_wave_barrier();
  }
}

extern "C" void kernel_launch(void* const* d_in, const int* in_sizes, int n_in,
                              void* d_out, int out_size, void* d_ws,
                              size_t ws_size, hipStream_t stream) {
  (void)in_sizes; (void)n_in; (void)out_size; (void)ws_size;
  const float* x         = (const float*)d_in[0];
  const int*   src       = (const int*)d_in[1];
  const int*   dst       = (const int*)d_in[2];
  const float* edge_attr = (const float*)d_in[3];
  const float* gcn_w     = (const float*)d_in[4];
  const float* gcn_b     = (const float*)d_in[5];
  const float* edge_w0   = (const float*)d_in[6];
  const float* edge_b0   = (const float*)d_in[7];
  const float* edge_w    = (const float*)d_in[8];
  const float* edge_b    = (const float*)d_in[9];
  const float* out_w0    = (const float*)d_in[10];
  const float* out_b0    = (const float*)d_in[11];
  const float* out_w     = (const float*)d_in[12];
  const float* out_b     = (const float*)d_in[13];
  const float* fin_w     = (const float*)d_in[14];
  const float* fin_b     = (const float*)d_in[15];
  float* out = (float*)d_out;

  char* ws = (char*)d_ws;
  const size_t OFF_XL   = 262144;                  // dinv in [0, 200000)
  const size_t OFF_AGG  = OFF_XL + 12800000;
  const size_t OFF_XC   = OFF_AGG + 12800000;
  const size_t OFF_PACK = OFF_XC + 12800000;
  const size_t OFF_XH   = OFF_PACK + 262144;       // packed uses 143360 B
  float*    dinv   = (float*)ws;
  float*    xl     = (float*)(ws + OFF_XL);
  float*    agg    = (float*)(ws + OFF_AGG);
  float*    xc     = (float*)(ws + OFF_XC);
  _Float16* packed = (_Float16*)(ws + OFF_PACK);
  _Float16* xh     = (_Float16*)(ws + OFF_XH);     // 6.4 MB

  k_pack<<<280, 256, 0, stream>>>(gcn_w, edge_w0, edge_w, out_w0, out_w, packed);
  hipMemsetAsync(dinv, 0, NNODES * sizeof(float), stream);
  k_deg<<<NEDGES / 256, 256, 0, stream>>>(dst, dinv);
  k_rsq<<<(NNODES + 255) / 256, 256, 0, stream>>>(dinv);

  for (int l = 0; l < 4; ++l) {
    const float* xin = (l == 0) ? x : xc;
    k_gcn_gemm<<<200, 256, 0, stream>>>(xin, packed + (size_t)l * 8 * 512, xl);
    hipMemsetAsync(agg, 0, (size_t)NNODES * 64 * sizeof(float), stream);
    k_scatter<<<(NEDGES * 64) / 256, 256, 0, stream>>>(src, dst, dinv, xl, agg);
    k_update<<<(NNODES * 64) / 256, 256, 0, stream>>>(agg, xl, dinv,
                                                      gcn_b + l * 64, xc);
  }

  k_cvt_x<<<(NNODES * 64) / 256, 256, 0, stream>>>(xc, xh);

  const size_t lds = (size_t)(N_EDGE_FRAGS * 512 + 8 * 1152) * sizeof(_Float16);
  k_edge_fused<<<512, 256, lds, stream>>>(
      xh, src, dst, edge_attr, packed + (size_t)EDGE_FRAG_BASE * 512,
      edge_b0, edge_b, out_b0, out_b, fin_w, fin_b, out);
}
